// NRDELayer_72765335929664
// MI455X (gfx1250) — matrix operand order repaired and checked by reading, not compiled, not run on hardware
//
#include <hip/hip_runtime.h>
#include <hip/hip_bf16.h>

// NRDE layer, Heun (RK2) integrator, restructured so the dominant contraction
//   out[b,h] = sum_w act[b,w] * ( sum_d ml_w[w,(h,d)] * coeff[b,d] )
// is computed as one big bf16 WMMA GEMM per step (M_j tensor), halving FLOPs
// vs the naive 2x big-GEMM-per-step form. Everything stays resident in L2.
//
// B=128, S=64, L=137, H=128, D_IN=64, W=256.

typedef __attribute__((ext_vector_type(16))) __bf16 v16bf;
typedef __attribute__((ext_vector_type(8)))  __bf16 v8bf;
typedef __attribute__((ext_vector_type(8)))  float  v8f;

#define B_   128
#define S_   64
#define L_   137
#define H_   128
#define DIN_ 64
#define W_   256
#define DM_  136   // L-1
#define DP_  160   // DM padded to multiple of 32 (bf16 WMMA K)

// ---------------------------------------------------------------------------
// WMMA helpers (layouts per CDNA5 ISA 7.12.2, wave32)
// ---------------------------------------------------------------------------
__device__ __forceinline__ v16bf cat8(v8bf lo, v8bf hi) {
  return __builtin_shufflevector(lo, hi, 0,1,2,3,4,5,6,7,8,9,10,11,12,13,14,15);
}

__device__ __forceinline__ v8f wmma_bf16(v16bf a, v16bf b, v8f c) {
  return __builtin_amdgcn_wmma_f32_16x16x32_bf16(false, a, false, b, (short)0, c,
                                                 false, false);
}

// A fragment, 16x32 bf16, source row-major; `row` points at A[m][0].
// Lane layout: element e<8 -> k = khalf*8+e ; e>=8 -> k = 16+khalf*8+(e-8).
__device__ __forceinline__ v16bf fragA_bf16(const __bf16* __restrict__ row,
                                            int kbase, int khalf) {
  const __bf16* p = row + kbase + khalf * 8;
  v8bf lo = *(const v8bf*)(p);
  v8bf hi = *(const v8bf*)(p + 16);
  return cat8(lo, hi);
}

// B fragment, 32x16 bf16; `col` points at Wt[n][0] (k-contiguous layout).
// Lane layout: element e<8 -> k = khalf*8+e ; e>=8 -> k = 16+khalf*8+(e-8).
__device__ __forceinline__ v16bf fragB_bf16(const __bf16* __restrict__ col,
                                            int kbase, int khalf) {
  const __bf16* p = col + kbase + khalf * 8;
  v8bf lo = *(const v8bf*)(p);
  v8bf hi = *(const v8bf*)(p + 16);
  return cat8(lo, hi);
}

// A fragment from f32 activations (convert to bf16 in-register); optionally
// fuses the Heun midpoint  x = y + alpha * k1.
template <bool MID>
__device__ __forceinline__ v16bf fragA_f32(const float* __restrict__ row,
                                           const float* __restrict__ row2,
                                           float alpha, int kbase, int khalf) {
  const float* p  = row + kbase + khalf * 8;
  const float* p2 = MID ? (row2 + kbase + khalf * 8) : nullptr;
  v16bf a;
#pragma unroll
  for (int e = 0; e < 8; ++e) {
    float x = p[e];
    if (MID) x = fmaf(alpha, p2[e], x);
    a[e] = (__bf16)x;
  }
#pragma unroll
  for (int e = 0; e < 8; ++e) {
    float x = p[16 + e];
    if (MID) x = fmaf(alpha, p2[16 + e], x);
    a[8 + e] = (__bf16)x;
  }
  return a;
}

// ---------------------------------------------------------------------------
// Prep kernels (run once per launch; all ws reads are preceded by writes)
// ---------------------------------------------------------------------------

// Transpose f32 weight [K][N] -> bf16 [N][K] (k-contiguous for B fragments).
__global__ __launch_bounds__(256) void prep_wT(const float* __restrict__ src,
                                               __bf16* __restrict__ dst,
                                               int K, int N) {
  int i = blockIdx.x * blockDim.x + threadIdx.x;
  if (i >= K * N) return;
  int n = i / K, k = i - n * K;
  dst[i] = (__bf16)src[(size_t)k * N + n];
}

// ml_w (W x H*DM) -> mlwp[h][w][dp] bf16, d zero-padded to DP.
__global__ __launch_bounds__(256) void prep_mlwp(const float* __restrict__ ml_w,
                                                 __bf16* __restrict__ mlwp) {
  size_t i = (size_t)blockIdx.x * blockDim.x + threadIdx.x;
  if (i >= (size_t)H_ * W_ * DP_) return;
  int dp = (int)(i % DP_);
  size_t t = i / DP_;
  int w = (int)(t % W_);
  int h = (int)(t / W_);
  mlwp[i] = (dp < DM_) ? (__bf16)ml_w[(size_t)w * (H_ * DM_) + h * DM_ + dp]
                       : (__bf16)0.0f;
}

// logsig (B,S,L) -> coeffp[j][b][dp] bf16 (coeff = logsig[...,1:], padded).
__global__ __launch_bounds__(256) void prep_coeffp(const float* __restrict__ logsig,
                                                   __bf16* __restrict__ coeffp) {
  size_t i = (size_t)blockIdx.x * blockDim.x + threadIdx.x;
  if (i >= (size_t)S_ * B_ * DP_) return;
  int dp = (int)(i % DP_);
  size_t t = i / DP_;
  int b = (int)(t % B_);
  int j = (int)(t / B_);
  coeffp[i] = (dp < DM_)
                  ? (__bf16)logsig[((size_t)b * S_ + j) * L_ + 1 + dp]
                  : (__bf16)0.0f;
}

// mlb[j][b][h] = sum_d ml_b[h*DM+d] * coeff[b,j,d]   (once, f32)
__global__ __launch_bounds__(128) void mlb_kernel(const float* __restrict__ logsig,
                                                  const float* __restrict__ ml_b,
                                                  float* __restrict__ mlb) {
  int j = blockIdx.x, b = blockIdx.y, h = threadIdx.x;
  const float* ls = logsig + ((size_t)b * S_ + j) * L_ + 1;
  const float* mb = ml_b + (size_t)h * DM_;
  float acc = 0.0f;
#pragma unroll 8
  for (int d = 0; d < DM_; ++d) acc = fmaf(mb[d], ls[d], acc);
  mlb[((size_t)j * B_ + b) * H_ + h] = acc;
}

// h0 = initial @ in_w + in_b  ->  ybuf and features[:,0,:]
__global__ __launch_bounds__(256) void h0_kernel(const float* __restrict__ initial,
                                                 const float* __restrict__ in_w,
                                                 const float* __restrict__ in_b,
                                                 float* __restrict__ ybuf,
                                                 float* __restrict__ features) {
  int i = blockIdx.x * blockDim.x + threadIdx.x;  // 0..B*H-1
  int b = i >> 7, h = i & (H_ - 1);
  float acc = in_b[h];
#pragma unroll 8
  for (int k = 0; k < DIN_; ++k)
    acc = fmaf(initial[b * DIN_ + k], in_w[k * H_ + h], acc);
  ybuf[i] = acc;
  features[(size_t)b * (S_ + 1) * H_ + h] = acc;
}

// ---------------------------------------------------------------------------
// M_j GEMM (dominant WMMA kernel): for each h,
//   M[w,b] = mlwp_h (256 x 160) @ coeffp_j^T (160 x 128)
// stored as Mbuf[b][h][w] bf16 (w-contiguous for the vec-mat consumer).
// One wave per block; each wave: 2 w-tiles x 4 b-tiles, 5 K-steps -> 40 WMMA.
// Per K-step: 2 A-frag + 4 B-frag loads feed 8 WMMAs (1.5 b128 loads/WMMA).
// ---------------------------------------------------------------------------
__global__ __launch_bounds__(32) void gemm_M_kernel(
    const __bf16* __restrict__ mlwp,    // [H][W][DP]
    const __bf16* __restrict__ coeffp,  // [S][B][DP]
    int j, __bf16* __restrict__ Mbuf) { // [B][H][W]
  const int lane = threadIdx.x;
  const int nloc = lane & 15;
  const int khalf = lane >> 4;   // also mhalf for C/D
  const int wp = blockIdx.x;     // 0..7   (pairs of w tiles)
  const int bg = blockIdx.y;     // 0..1   (groups of 4 b-tiles)
  const int h  = blockIdx.z;     // 0..127

  const __bf16* Arow0 =
      mlwp + ((size_t)h * W_ + (wp * 2 + 0) * 16 + nloc) * DP_;
  const __bf16* Arow1 =
      mlwp + ((size_t)h * W_ + (wp * 2 + 1) * 16 + nloc) * DP_;
  const __bf16* Bbase = coeffp + (size_t)j * B_ * DP_;
  const __bf16* Bcol[4];
#pragma unroll
  for (int t = 0; t < 4; ++t)
    Bcol[t] = Bbase + (size_t)(bg * 64 + t * 16 + nloc) * DP_;

  v8f acc0[4] = {};
  v8f acc1[4] = {};
#pragma unroll
  for (int kb = 0; kb < DP_; kb += 32) {
    v16bf a0 = fragA_bf16(Arow0, kb, khalf);
    v16bf a1 = fragA_bf16(Arow1, kb, khalf);
#pragma unroll
    for (int t = 0; t < 4; ++t) {
      v16bf b = fragB_bf16(Bcol[t], kb, khalf);
      acc0[t] = wmma_bf16(a0, b, acc0[t]);
      acc1[t] = wmma_bf16(a1, b, acc1[t]);
    }
  }
#pragma unroll
  for (int t = 0; t < 4; ++t) {
    const int b = bg * 64 + t * 16 + nloc;
    __bf16* out = Mbuf + ((size_t)b * H_ + h) * W_;
    v8bf d0, d1;
#pragma unroll
    for (int r = 0; r < 8; ++r) {
      d0[r] = (__bf16)acc0[t][r];
      d1[r] = (__bf16)acc1[t][r];
    }
    *(v8bf*)(out + (wp * 2 + 0) * 16 + khalf * 8) = d0;
    *(v8bf*)(out + (wp * 2 + 1) * 16 + khalf * 8) = d1;
  }
}

// ---------------------------------------------------------------------------
// MLP layer: Y(128xN=256) = act( X(128xK) @ Wt^T + bias ),  Wt is bf16 [N][K].
// MID fuses X = y + alpha*k1 with alpha = times[tn+1]-times[tn].
// One wave per block: 1 m-tile x 4 n-tiles (A f32->bf16 conversion amortized
// over 4 WMMAs).
// ---------------------------------------------------------------------------
template <int K, bool MID, int ACT>  // ACT: 0=relu, 1=tanh
__global__ __launch_bounds__(32) void mlp_kernel(
    const float* __restrict__ X, const float* __restrict__ X2,
    const float* __restrict__ times, int tn,
    const __bf16* __restrict__ Wt, const float* __restrict__ bias,
    float* __restrict__ Y) {
  const int lane = threadIdx.x;
  const int nloc = lane & 15;
  const int khalf = lane >> 4;
  const int mt = blockIdx.x;  // 0..7   batch tiles
  const int ng = blockIdx.y;  // 0..3   groups of 4 n tiles (N=256)

  float alpha = 0.0f;
  if (MID) alpha = times[tn + 1] - times[tn];

  const float* Arow  = X + (size_t)(mt * 16 + nloc) * K;
  const float* Arow2 = MID ? X2 + (size_t)(mt * 16 + nloc) * K : nullptr;
  const __bf16* Bcol[4];
#pragma unroll
  for (int t = 0; t < 4; ++t)
    Bcol[t] = Wt + (size_t)(ng * 64 + t * 16 + nloc) * K;

  v8f acc[4] = {};
#pragma unroll
  for (int kb = 0; kb < K; kb += 32) {
    v16bf a = fragA_f32<MID>(Arow, Arow2, alpha, kb, khalf);
#pragma unroll
    for (int t = 0; t < 4; ++t)
      acc[t] = wmma_bf16(a, fragB_bf16(Bcol[t], kb, khalf), acc[t]);
  }
#pragma unroll
  for (int t = 0; t < 4; ++t) {
    const int n = ng * 64 + t * 16 + nloc;
    const float bn = bias[n];
#pragma unroll
    for (int r = 0; r < 8; ++r) {
      int m = mt * 16 + khalf * 8 + r;
      float v = acc[t][r] + bn;
      v = (ACT == 0) ? fmaxf(v, 0.0f) : tanhf(v);
      Y[(size_t)m * W_ + n] = v;
    }
  }
}

// ---------------------------------------------------------------------------
// Vector-matrix contraction per batch:
//   kout[b,h] = ( sum_w act[b,w]*M[b][h][w] + mlb_j[b,h] ) / dt
// dt = times[idx]-times[idx-1] read on device.
// ---------------------------------------------------------------------------
__global__ __launch_bounds__(128) void vm_kernel(
    const float* __restrict__ act,    // [B][W]
    const __bf16* __restrict__ Mbuf,  // [B][H][W]
    const float* __restrict__ mlbj,   // [B][H]  (slice for this j)
    const float* __restrict__ times, int idx,
    float* __restrict__ kout) {
  const int b = blockIdx.x, h = threadIdx.x;
  __shared__ float sact[W_];
  for (int i = threadIdx.x; i < W_; i += blockDim.x) sact[i] = act[b * W_ + i];
  __syncthreads();
  const __bf16* Mr = Mbuf + ((size_t)b * H_ + h) * W_;
  float acc = 0.0f;
#pragma unroll 4
  for (int w = 0; w < W_; w += 8) {
    v8bf mv = *(const v8bf*)(Mr + w);
#pragma unroll
    for (int e = 0; e < 8; ++e) acc = fmaf(sact[w + e], (float)mv[e], acc);
  }
  const float dt = times[idx] - times[idx - 1];
  kout[(size_t)b * H_ + h] = (acc + mlbj[(size_t)b * H_ + h]) / dt;
}

// Heun update: y1 = y + 0.5*dt*(k1+k2); writes ybuf, features[:,n+1,:], last.
__global__ __launch_bounds__(256) void update_kernel(
    float* __restrict__ ybuf, const float* __restrict__ k1,
    const float* __restrict__ k2, const float* __restrict__ times, int n,
    float* __restrict__ features, float* __restrict__ lastout) {
  int i = blockIdx.x * blockDim.x + threadIdx.x;  // 0..B*H-1
  const float dt = times[n + 1] - times[n];
  const float y1 = ybuf[i] + 0.5f * dt * (k1[i] + k2[i]);
  ybuf[i] = y1;
  int b = i >> 7, h = i & (H_ - 1);
  features[((size_t)b * (S_ + 1) + (n + 1)) * H_ + h] = y1;
  if (n == S_ - 1) lastout[i] = y1;
}

// ---------------------------------------------------------------------------
// Host launcher
// ---------------------------------------------------------------------------
extern "C" void kernel_launch(void* const* d_in, const int* in_sizes, int n_in,
                              void* d_out, int out_size, void* d_ws,
                              size_t ws_size, hipStream_t stream) {
  const float* times   = (const float*)d_in[0];
  const float* logsig  = (const float*)d_in[1];
  const float* initial = (const float*)d_in[2];
  const float* in_w    = (const float*)d_in[3];
  const float* in_b    = (const float*)d_in[4];
  const float* h1_w    = (const float*)d_in[5];
  const float* h1_b    = (const float*)d_in[6];
  const float* h2_w    = (const float*)d_in[7];
  const float* h2_b    = (const float*)d_in[8];
  const float* vo_w    = (const float*)d_in[9];
  const float* vo_b    = (const float*)d_in[10];
  const float* ml_w    = (const float*)d_in[11];
  const float* ml_b    = (const float*)d_in[12];

  float* features = (float*)d_out;                          // B x (S+1) x H
  float* lastout  = features + (size_t)B_ * (S_ + 1) * H_;  // B x H

  // Workspace carve (~35 MB total), 256B-aligned slices.
  char* wp = (char*)d_ws;
  auto carve = [&](size_t bytes) -> void* {
    void* p = (void*)wp;
    wp += (bytes + 255) & ~(size_t)255;
    return p;
  };
  __bf16* h1wb   = (__bf16*)carve((size_t)W_ * H_ * 2);           // [256][128]
  __bf16* h2wb   = (__bf16*)carve((size_t)W_ * W_ * 2);           // [256][256]
  __bf16* vowb   = (__bf16*)carve((size_t)W_ * W_ * 2);           // [256][256]
  __bf16* mlwp   = (__bf16*)carve((size_t)H_ * W_ * DP_ * 2);     // 10.5 MB
  __bf16* coeffp = (__bf16*)carve((size_t)S_ * B_ * DP_ * 2);     // 2.6 MB
  float*  mlb    = (float*) carve((size_t)S_ * B_ * H_ * 4);      // 4.2 MB
  __bf16* Mbuf0  = (__bf16*)carve((size_t)B_ * H_ * W_ * 2);      // 8.4 MB
  __bf16* Mbuf1  = (__bf16*)carve((size_t)B_ * H_ * W_ * 2);      // 8.4 MB
  float*  ybuf   = (float*) carve((size_t)B_ * H_ * 4);
  float*  k1buf  = (float*) carve((size_t)B_ * H_ * 4);
  float*  k2buf  = (float*) carve((size_t)B_ * H_ * 4);
  float*  act1   = (float*) carve((size_t)B_ * W_ * 4);
  float*  act2   = (float*) carve((size_t)B_ * W_ * 4);
  float*  hact   = (float*) carve((size_t)B_ * W_ * 4);
  __bf16* Mb[2] = {Mbuf0, Mbuf1};

  // --- one-time prep ---
  prep_wT<<<(H_ * W_ + 255) / 256, 256, 0, stream>>>(h1_w, h1wb, H_, W_);
  prep_wT<<<(W_ * W_ + 255) / 256, 256, 0, stream>>>(h2_w, h2wb, W_, W_);
  prep_wT<<<(W_ * W_ + 255) / 256, 256, 0, stream>>>(vo_w, vowb, W_, W_);
  {
    size_t n = (size_t)H_ * W_ * DP_;
    prep_mlwp<<<(unsigned)((n + 255) / 256), 256, 0, stream>>>(ml_w, mlwp);
  }
  {
    size_t n = (size_t)S_ * B_ * DP_;
    prep_coeffp<<<(unsigned)((n + 255) / 256), 256, 0, stream>>>(logsig, coeffp);
  }
  mlb_kernel<<<dim3(S_, B_), 128, 0, stream>>>(logsig, ml_b, mlb);
  h0_kernel<<<(B_ * H_) / 256, 256, 0, stream>>>(initial, in_w, in_b, ybuf,
                                                 features);
  // M_0 (used by both k1 and k2 at step 0, and by k1 at step 1)
  gemm_M_kernel<<<dim3(8, 2, H_), 32, 0, stream>>>(mlwp, coeffp, 0, Mb[0]);

  const dim3 mgrid(8, 4);
  for (int n = 0; n < S_; ++n) {
    const int j1 = (n == 0) ? 0 : n - 1;   // coeff index for k1
    const int i1 = (n == 0) ? 1 : n;       // dt index for k1
    const int j2 = n, i2 = n + 1;          // coeff/dt index for k2
    if (n > 0)  // compute M_n for k2 (and k1 of step n+1); ping-pong buffers
      gemm_M_kernel<<<dim3(8, 2, H_), 32, 0, stream>>>(mlwp, coeffp, n,
                                                       Mb[n & 1]);
    // k1 = func(t0, y)
    mlp_kernel<H_, false, 0><<<mgrid, 32, 0, stream>>>(ybuf, nullptr, times, 0,
                                                       h1wb, h1_b, act1);
    mlp_kernel<W_, false, 0><<<mgrid, 32, 0, stream>>>(act1, nullptr, times, 0,
                                                       h2wb, h2_b, act2);
    mlp_kernel<W_, false, 1><<<mgrid, 32, 0, stream>>>(act2, nullptr, times, 0,
                                                       vowb, vo_b, hact);
    vm_kernel<<<B_, 128, 0, stream>>>(hact, Mb[j1 & 1],
                                      mlb + (size_t)j1 * B_ * H_, times, i1,
                                      k1buf);
    // k2 = func(t1, y + dt*k1)   (midpoint fused into first layer)
    mlp_kernel<H_, true, 0><<<mgrid, 32, 0, stream>>>(ybuf, k1buf, times, n,
                                                      h1wb, h1_b, act1);
    mlp_kernel<W_, false, 0><<<mgrid, 32, 0, stream>>>(act1, nullptr, times, 0,
                                                       h2wb, h2_b, act2);
    mlp_kernel<W_, false, 1><<<mgrid, 32, 0, stream>>>(act2, nullptr, times, 0,
                                                       vowb, vo_b, hact);
    vm_kernel<<<B_, 128, 0, stream>>>(hact, Mb[j2 & 1],
                                      mlb + (size_t)j2 * B_ * H_, times, i2,
                                      k2buf);
    // y_{n+1}
    update_kernel<<<(B_ * H_) / 256, 256, 0, stream>>>(ybuf, k1buf, k2buf,
                                                       times, n, features,
                                                       lastout);
  }
}
